// SemiSupervisedGCN_43499428774647
// MI455X (gfx1250) — matrix operand order repaired and checked by reading, not compiled, not run on hardware
//
#include <hip/hip_runtime.h>
#include <hip/hip_bf16.h>

typedef __attribute__((ext_vector_type(16))) __bf16       v16bf;
typedef __attribute__((ext_vector_type(8)))  float        v8f;
typedef __attribute__((ext_vector_type(4)))  float        f32x4;
typedef __attribute__((ext_vector_type(4)))  unsigned int u32x4;

#define D 128

__device__ __forceinline__ unsigned short f2bf(float f) {
  union { float f; unsigned int u; } x; x.f = f;
  unsigned int u = x.u;
  u += 0x7FFFu + ((u >> 16) & 1u);   // round-to-nearest-even
  return (unsigned short)(u >> 16);
}

// ---------------- zero (float4 stores) ----------------
__global__ void zero_kernel(float* __restrict__ p, unsigned n4) {
  unsigned i = blockIdx.x * blockDim.x + threadIdx.x;
  if (i < n4) {
    f32x4 z = {0.f, 0.f, 0.f, 0.f};
    ((f32x4*)p)[i] = z;
  }
}

// ---------------- edge scatter: one wave per edge ----------------
__global__ void scatter_kernel(const float* __restrict__ x,
                               const int* __restrict__ src,
                               const int* __restrict__ dst,
                               const float* __restrict__ ew,
                               float* __restrict__ agg, int E) {
  int e    = (int)((blockIdx.x * (unsigned)blockDim.x + threadIdx.x) >> 5);
  int lane = threadIdx.x & 31;
  if (e >= E) return;
  int s = src[e], d = dst[e];
  float w = ew[e];
  f32x4 v = ((const f32x4*)x)[(unsigned)s * 32u + lane];   // coalesced 512B/edge
  float* ap = agg + ((unsigned)d * (unsigned)D + (unsigned)(lane * 4));
  atomicAdd(ap + 0, v[0] * w);
  atomicAdd(ap + 1, v[1] * w);
  atomicAdd(ap + 2, v[2] * w);
  atomicAdd(ap + 3, v[3] * w);
}

// ---------------- WMMA GEMM: out = relu((A1 [+ A2]) @ W + bias) ----------------
// A1/A2: [nrows x 128] f32, W: [128 x NOUT] f32 row-major, out: [nrows x NOUT] f32
template <int NOUT>
__global__ void __launch_bounds__(256)
gemm_bias_relu(const float* __restrict__ A1, const float* __restrict__ A2,
               const float* __restrict__ W, const float* __restrict__ bias,
               float* __restrict__ out, int nrows) {
  constexpr int LDA = D + 8;      // ushort stride: 272B rows -> conflict-free col reads
  constexpr int LDW = D + 8;
  __shared__ __attribute__((aligned(16))) unsigned short Asm[128 * LDA];
  __shared__ __attribute__((aligned(16))) unsigned short Wt[NOUT * LDW];

  const int tid = threadIdx.x;
  const int rowbase = (int)blockIdx.x * 128;
  const bool full = (rowbase + 128) <= nrows;   // block-uniform

  // stage A = A1 (+ A2) -> bf16 LDS
  {
    int r  = tid >> 1;
    int cb = (tid & 1) * 64;
    int grow = rowbase + r;
    bool ok = grow < nrows;
    unsigned gro = (unsigned)grow * (unsigned)D + (unsigned)cb;
    const f32x4* p1 = (const f32x4*)(A1 + gro);
    const f32x4* p2 = (const f32x4*)(A2 ? (A2 + gro) : nullptr);
#pragma unroll
    for (int j = 0; j < 16; ++j) {
      f32x4 v = {0.f, 0.f, 0.f, 0.f};
      if (ok) {
        v = p1[j];
        if (A2) { f32x4 v2 = p2[j]; v = v + v2; }
      }
      int c = cb + j * 4;
      Asm[r * LDA + c + 0] = f2bf(v[0]);
      Asm[r * LDA + c + 1] = f2bf(v[1]);
      Asm[r * LDA + c + 2] = f2bf(v[2]);
      Asm[r * LDA + c + 3] = f2bf(v[3]);
    }
  }
  // stage W transposed -> bf16 LDS (Wt[n][k])
  for (int i = tid * 4; i < D * NOUT; i += 256 * 4) {
    int k = i / NOUT, n = i % NOUT;
    f32x4 v = *(const f32x4*)(W + i);
    Wt[(n + 0) * LDW + k] = f2bf(v[0]);
    Wt[(n + 1) * LDW + k] = f2bf(v[1]);
    Wt[(n + 2) * LDW + k] = f2bf(v[2]);
    Wt[(n + 3) * LDW + k] = f2bf(v[3]);
  }
  __syncthreads();

  constexpr int NT = NOUT / 16;   // column tiles
  constexpr int RG = 8 / NT;      // row groups of waves
  constexpr int MT = 8 / RG;      // m-tiles per wave
  const int wave = tid >> 5, lane = tid & 31;
  const int ct = wave % NT;
  const int rg = wave / NT;
  const int nlo = lane & 15;
  const int hi  = lane >> 4;
  const int ncol = ct * 16 + nlo;

  union Frag { v16bf v; u32x4 q[2]; };

  // B fragments for all 4 k-steps (B 32x16 bf16: lane group hi -> K = hi*8..+7 and 16+hi*8..+7)
  Frag bf[4];
#pragma unroll
  for (int kt = 0; kt < 4; ++kt) {
    int ks = kt * 32 + hi * 8;
    const u32x4* p = (const u32x4*)&Wt[ncol * LDW + ks];
    bf[kt].q[0] = p[0];
    bf[kt].q[1] = p[2];
  }
  float bv = bias[ncol];

#pragma unroll
  for (int m = 0; m < MT; ++m) {
    int mt = rg * MT + m;
    int rloc = mt * 16 + nlo;       // A row index in LDS for this lane
    v8f acc = {0.f, 0.f, 0.f, 0.f, 0.f, 0.f, 0.f, 0.f};
#pragma unroll
    for (int kt = 0; kt < 4; ++kt) {
      int kb = kt * 32 + hi * 8;    // A 16x32 bf16: K = kb..kb+7 and kb+16..kb+23
      Frag a;
      a.q[0] = *(const u32x4*)&Asm[rloc * LDA + kb];
      a.q[1] = *(const u32x4*)&Asm[rloc * LDA + kb + 16];
      acc = __builtin_amdgcn_wmma_f32_16x16x32_bf16(
          false, a.v, false, bf[kt].v, (short)0, acc, false, false);
    }
    // epilogue: D layout -> lane holds col ncol, rows (hi*8 + r) of tile mt
    int gr0 = rowbase + mt * 16 + hi * 8;
    float* po = out + ((unsigned)gr0 * (unsigned)NOUT + (unsigned)ncol);
    if (full) {
#pragma unroll
      for (int r = 0; r < 8; ++r) {
        float v = acc[r] + bv;
        po[r * NOUT] = v > 0.f ? v : 0.f;     // saddr + imm-offset stores
      }
    } else {
#pragma unroll
      for (int r = 0; r < 8; ++r) {
        if (gr0 + r < nrows) {
          float v = acc[r] + bv;
          po[r * NOUT] = v > 0.f ? v : 0.f;
        }
      }
    }
  }
}

// ---------------- final head: logits = h @ Wc2 + bc2 (64 -> 2) ----------------
__global__ void head_kernel(const float* __restrict__ h, const float* __restrict__ Wc2,
                            const float* __restrict__ bc2, float* __restrict__ out,
                            int nrows) {
  __shared__ float ws[130];
  if (threadIdx.x < 128) ws[threadIdx.x] = Wc2[threadIdx.x];
  if (threadIdx.x < 2)   ws[128 + threadIdx.x] = bc2[threadIdx.x];
  __syncthreads();
  int row = (int)(blockIdx.x * blockDim.x + threadIdx.x);
  if (row >= nrows) return;
  const f32x4* hp = (const f32x4*)h + (unsigned)row * 16u;
  float l0 = ws[128], l1 = ws[129];
#pragma unroll
  for (int i = 0; i < 16; ++i) {
    f32x4 v = hp[i];
#pragma unroll
    for (int j = 0; j < 4; ++j) {
      int c = i * 4 + j;
      l0 += v[j] * ws[2 * c + 0];
      l1 += v[j] * ws[2 * c + 1];
    }
  }
  out[(unsigned)row * 2u + 0u] = l0;
  out[(unsigned)row * 2u + 1u] = l1;
}

extern "C" void kernel_launch(void* const* d_in, const int* in_sizes, int n_in,
                              void* d_out, int out_size, void* d_ws, size_t ws_size,
                              hipStream_t stream) {
  const float* feat = (const float*)d_in[0];
  const int*   eidx = (const int*)d_in[1];
  const float* ew   = (const float*)d_in[2];
  const float* W1   = (const float*)d_in[3];
  const float* b1   = (const float*)d_in[4];
  const float* W2   = (const float*)d_in[5];
  const float* b2   = (const float*)d_in[6];
  const float* Wc1  = (const float*)d_in[7];
  const float* bc1  = (const float*)d_in[8];
  const float* Wc2  = (const float*)d_in[9];
  const float* bc2  = (const float*)d_in[10];

  const int N = in_sizes[0] / D;
  const int E = in_sizes[1] / 2;
  const int* src = eidx;
  const int* dst = eidx + E;

  float* agg = (float*)d_ws;                 // N*128 f32
  float* x1  = agg + (size_t)N * D;          // N*128 f32
  float* h   = x1;                           // overlay: x1 dead before h is written

  unsigned n4 = (unsigned)N * D / 4;
  int zblocks = (int)((n4 + 255) / 256);
  int sblocks = (E + 7) / 8;                 // 8 waves (edges) per 256-thread block
  int gblocks = (N + 127) / 128;
  int hblocks = (N + 255) / 256;

  // layer 1: agg1 = scatter(feat); x1 = relu((agg1+feat)@W1 + b1)
  zero_kernel<<<zblocks, 256, 0, stream>>>(agg, n4);
  scatter_kernel<<<sblocks, 256, 0, stream>>>(feat, src, dst, ew, agg, E);
  gemm_bias_relu<128><<<gblocks, 256, 0, stream>>>(agg, feat, W1, b1, x1, N);

  // layer 2: agg2 = scatter(x1); x2 = relu((agg2+x1)@W2 + b2) (in-place into agg)
  zero_kernel<<<zblocks, 256, 0, stream>>>(agg, n4);
  scatter_kernel<<<sblocks, 256, 0, stream>>>(x1, src, dst, ew, agg, E);
  gemm_bias_relu<128><<<gblocks, 256, 0, stream>>>(agg, x1, W2, b2, agg, N);

  // classifier: h = relu(x2 @ Wc1 + bc1); logits = h @ Wc2 + bc2
  gemm_bias_relu<64><<<gblocks, 256, 0, stream>>>(agg, nullptr, Wc1, bc1, h, N);
  head_kernel<<<hblocks, 256, 0, stream>>>(h, Wc2, bc2, (float*)d_out, N);
}
